// TransformerBlock_28827820491030
// MI455X (gfx1250) — matrix operand order repaired and checked by reading, not compiled, not run on hardware
//
#include <hip/hip_runtime.h>
#include <hip/hip_bf16.h>

// ---------------------------------------------------------------------------
// Transformer block for MI455X (gfx1250), wave32 + WMMA bf16 (f32 accumulate)
// B=2, S=2048, D=256, H=16, INNER=4096
// GEMM waves compute 32x64 tiles; K-loop unrolled x2 with two named fragment
// buffer sets (no register copies -> no WMMA hazard v_nops in the hot loop).
// ---------------------------------------------------------------------------

#define B_     2
#define S_     2048
#define D_     256
#define H_     16
#define INNER_ 4096
#define BS_    (B_ * S_)

typedef __attribute__((ext_vector_type(16))) __bf16 v16bf;
typedef __attribute__((ext_vector_type(8)))  float  v8f;

union V16U { v16bf v; uint4 u[2]; };

// ---- raw fragment loads (two 16B chunks per lane) -------------------------
__device__ __forceinline__ v16bf load_pair16(const __bf16* p) {  // +0, +16
  V16U r;
  r.u[0] = *reinterpret_cast<const uint4*>(p);
  r.u[1] = *reinterpret_cast<const uint4*>(p + 16);
  return r.v;
}
__device__ __forceinline__ v16bf load_pair8(const __bf16* p) {   // +0, +16
  V16U r;
  r.u[0] = *reinterpret_cast<const uint4*>(p);
  r.u[1] = *reinterpret_cast<const uint4*>(p + 16);
  return r.v;
}

// ---- WMMA fragment loaders (wave32 layouts per CDNA5 ISA 7.12.2) ----------
// A fragment 16x32 bf16: lane = half*16 + m ; elem e<8 -> K = 8*half + e,
// elem e>=8 -> K = 16 + 8*half + (e-8).
__device__ __forceinline__ v16bf load_a_frag(const __bf16* A, int lda,
                                             int row0, int k0) {
  int lane = threadIdx.x & 31;
  int m = lane & 15, half = lane >> 4;
  return load_pair16(A + (size_t)(row0 + m) * lda + k0 + 8 * half);
}

// B fragment 32x16 bf16 from a [N,K] (transposed) matrix so each lane reads
// two 16B chunks: lane = khalf*16 + n ; elem e<8 -> K = 8*khalf + e, e>=8 -> K = 16 + 8*khalf + (e-8).
__device__ __forceinline__ v16bf load_bt_frag(const __bf16* Bt, int ldk,
                                              int col0, int k0) {
  int lane = threadIdx.x & 31;
  int n = lane & 15, kb = (lane >> 4) * 8;
  return load_pair8(Bt + (size_t)(col0 + n) * ldk + k0 + kb);
}

__device__ __forceinline__ v8f wmma_bf16(v16bf a, v16bf b, v8f c) {
  return __builtin_amdgcn_wmma_f32_16x16x32_bf16(
      false, a, false, b, (short)0, c, false, false);
}

// row-wise reductions across the 16 lanes that hold one C/D row
__device__ __forceinline__ float rowmax16(float v) {
  #pragma unroll
  for (int m = 1; m <= 8; m <<= 1) v = fmaxf(v, __shfl_xor(v, m, 32));
  return v;
}
__device__ __forceinline__ float rowsum16(float v) {
  #pragma unroll
  for (int m = 1; m <= 8; m <<= 1) v += __shfl_xor(v, m, 32);
  return v;
}

// ---- software-pipelined 32x64 GEMM core: acc += A[32,K] * Bt[64 cols of K]
// KDIM must be a multiple of 64. Double-buffered via two named fragment sets.
template <int KDIM>
__device__ __forceinline__ void gemm32x64(const __bf16* __restrict__ A, int lda,
                                          int row0,
                                          const __bf16* __restrict__ Bt,
                                          int col0, v8f acc[2][4]) {
  int lane = threadIdx.x & 31;
  int m = lane & 15, half = lane >> 4;
  int kb = half * 8;
  const __bf16* pa0 = A + (size_t)(row0 + m) * lda + 8 * half;
  const __bf16* pa1 = pa0 + (size_t)16 * lda;
  const __bf16* pb  = Bt + (size_t)(col0 + m) * KDIM + kb;  // n == m (lane&15)

  v16bf aA0 = load_pair16(pa0), aA1 = load_pair16(pa1);
  v16bf aB0, aB1, bA[4], bB[4];
  #pragma unroll
  for (int t = 0; t < 4; ++t) bA[t] = load_pair8(pb + (size_t)(16 * t) * KDIM);

  for (int k0 = 0; k0 < KDIM - 64; k0 += 64) {
    // far prefetch (global_prefetch_b8)
    __builtin_prefetch(pa0 + k0 + 128, 0, 0);
    __builtin_prefetch(pb + k0 + 128, 0, 0);
    // stage B loads (k0+32), then stage A math (k0)
    aB0 = load_pair16(pa0 + k0 + 32);
    aB1 = load_pair16(pa1 + k0 + 32);
    #pragma unroll
    for (int t = 0; t < 4; ++t)
      bB[t] = load_pair8(pb + (size_t)(16 * t) * KDIM + k0 + 32);
    #pragma unroll
    for (int t = 0; t < 4; ++t) {
      acc[0][t] = wmma_bf16(aA0, bA[t], acc[0][t]);
      acc[1][t] = wmma_bf16(aA1, bA[t], acc[1][t]);
    }
    // stage A loads (k0+64), then stage B math (k0+32)
    aA0 = load_pair16(pa0 + k0 + 64);
    aA1 = load_pair16(pa1 + k0 + 64);
    #pragma unroll
    for (int t = 0; t < 4; ++t)
      bA[t] = load_pair8(pb + (size_t)(16 * t) * KDIM + k0 + 64);
    #pragma unroll
    for (int t = 0; t < 4; ++t) {
      acc[0][t] = wmma_bf16(aB0, bB[t], acc[0][t]);
      acc[1][t] = wmma_bf16(aB1, bB[t], acc[1][t]);
    }
  }
  // tail: aA/bA hold K-step KDIM-64; load and run the final step KDIM-32
  aB0 = load_pair16(pa0 + KDIM - 32);
  aB1 = load_pair16(pa1 + KDIM - 32);
  #pragma unroll
  for (int t = 0; t < 4; ++t)
    bB[t] = load_pair8(pb + (size_t)(16 * t) * KDIM + KDIM - 32);
  #pragma unroll
  for (int t = 0; t < 4; ++t) {
    acc[0][t] = wmma_bf16(aA0, bA[t], acc[0][t]);
    acc[1][t] = wmma_bf16(aA1, bA[t], acc[1][t]);
  }
  #pragma unroll
  for (int t = 0; t < 4; ++t) {
    acc[0][t] = wmma_bf16(aB0, bB[t], acc[0][t]);
    acc[1][t] = wmma_bf16(aB1, bB[t], acc[1][t]);
  }
}

// ---------------------------------------------------------------------------
// 0) f32 -> bf16 transpose-convert:  dst[N,K] = src[K,N]
// ---------------------------------------------------------------------------
__global__ void k_convert_transpose(const float* __restrict__ src,
                                    __bf16* __restrict__ dst, int K, int N) {
  size_t idx = (size_t)blockIdx.x * blockDim.x + threadIdx.x;
  size_t total = (size_t)K * N;
  if (idx < total) {
    int k = (int)(idx / N), n = (int)(idx % N);
    dst[(size_t)n * K + k] = (__bf16)src[idx];
  }
}

// ---------------------------------------------------------------------------
// 1) LayerNorm over D_=256 columns, one block (256 thr) per row, -> bf16
// ---------------------------------------------------------------------------
__global__ void k_layernorm(const float* __restrict__ x,
                            const float* __restrict__ g,
                            const float* __restrict__ b,
                            __bf16* __restrict__ out) {
  __shared__ float red[D_];
  int row = blockIdx.x, t = threadIdx.x;
  float v = x[(size_t)row * D_ + t];
  red[t] = v; __syncthreads();
  #pragma unroll
  for (int s = D_ / 2; s > 0; s >>= 1) { if (t < s) red[t] += red[t + s]; __syncthreads(); }
  float mean = red[0] * (1.0f / D_); __syncthreads();
  float d = v - mean;
  red[t] = d * d; __syncthreads();
  #pragma unroll
  for (int s = D_ / 2; s > 0; s >>= 1) { if (t < s) red[t] += red[t + s]; __syncthreads(); }
  float inv = rsqrtf(red[0] * (1.0f / D_) + 1e-5f);
  out[(size_t)row * D_ + t] = (__bf16)(d * inv * g[t] + b[t]);
}

// ---------------------------------------------------------------------------
// 2) QKV projection + RoPE. grid (BS/32, INNER/512, 3), block 256 (8 waves).
//    Wave computes a 32x64 tile. q,k stored [B,H,S,D]; v stored transposed
//    [B,H,D,S] so attention's P@V B-fragments are contiguous.
// ---------------------------------------------------------------------------
__global__ __launch_bounds__(256) void k_qkv(
    const __bf16* __restrict__ hbf,
    const __bf16* __restrict__ Wqt, const __bf16* __restrict__ Wkt,
    const __bf16* __restrict__ Wvt,
    __bf16* __restrict__ qb, __bf16* __restrict__ kb,
    __bf16* __restrict__ vtb) {
  int z = blockIdx.z;
  const __bf16* Wt = (z == 0) ? Wqt : (z == 1) ? Wkt : Wvt;
  int wave = threadIdx.x >> 5;
  int lane = threadIdx.x & 31;
  int n = lane & 15, half = lane >> 4;
  int row0 = blockIdx.x * 32;
  int col0 = (blockIdx.y * 8 + wave) * 64;

  v8f acc[2][4] = {};
  gemm32x64<D_>(hbf, D_, row0, Wt, col0, acc);

  #pragma unroll
  for (int t = 0; t < 4; ++t) {
    int col = col0 + 16 * t + n;
    int head = col >> 8, d = col & 255;
    float invf = __expf(-(float)(d & ~1) * (9.210340372f / 256.0f)); // 10000^(-2i/D)
    #pragma unroll
    for (int g = 0; g < 2; ++g)
      #pragma unroll
      for (int r = 0; r < 8; ++r) {
        int row = row0 + g * 16 + r + 8 * half;
        int bb = row >> 11, s = row & (S_ - 1);
        float v = acc[g][t][r];
        if (z < 2) {                     // RoPE on q,k (pairs = adjacent lanes)
          float pv = __shfl_xor(v, 1, 32);
          float sn, cs; __sincosf((float)s * invf, &sn, &cs);
          v = (d & 1) ? (pv * sn + v * cs) : (v * cs - pv * sn);
          __bf16* dst = (z == 0) ? qb : kb;
          dst[(((size_t)bb * H_ + head) * S_ + s) * D_ + d] = (__bf16)v;
        } else {                         // V transposed: [b,h,d,s]
          vtb[(((size_t)bb * H_ + head) * D_ + d) * S_ + s] = (__bf16)v;
        }
      }
  }
}

// ---------------------------------------------------------------------------
// 3) Causal flash attention. One wave per 16-row q-tile; chunks of 32 keys.
//    grid (B*H, S/16/4), block 128 (4 waves). Output -> attn [B,S,INNER] bf16.
// ---------------------------------------------------------------------------
__global__ __launch_bounds__(128, 1) void k_attn(
    const __bf16* __restrict__ qb, const __bf16* __restrict__ kb,
    const __bf16* __restrict__ vtb, __bf16* __restrict__ attn) {
  __shared__ __bf16 plds_all[4][16 * 32];
  int bh = blockIdx.x;
  int wave = threadIdx.x >> 5;
  int lane = threadIdx.x & 31;
  int n = lane & 15, half = lane >> 4;
  int qt = blockIdx.y * 4 + wave;
  int q0 = qt * 16;
  __bf16* plds = plds_all[wave];

  const __bf16* Q  = qb  + (size_t)bh * S_ * D_;
  const __bf16* Kp = kb  + (size_t)bh * S_ * D_;
  const __bf16* Vt = vtb + (size_t)bh * D_ * S_;

  v16bf qf[8];
  #pragma unroll
  for (int i = 0; i < 8; ++i) qf[i] = load_a_frag(Q, D_, q0, i * 32);

  v8f acc[16] = {};
  float M[8], L[8];
  #pragma unroll
  for (int r = 0; r < 8; ++r) { M[r] = -3.0e38f; L[r] = 0.0f; }

  int cend = (q0 + 15) >> 5;               // inclusive last 32-key chunk
  for (int c = 0; c <= cend; ++c) {
    if (c < cend) {                        // prefetch next chunk's K/V rows
      __builtin_prefetch(Kp + (size_t)(c * 32 + 32 + n) * D_, 0, 0);
      __builtin_prefetch(Vt + (size_t)n * S_ + c * 32 + 32, 0, 0);
    }
    v8f sc[2];
    #pragma unroll
    for (int tt = 0; tt < 2; ++tt) {
      v8f s = {};
      #pragma unroll
      for (int i = 0; i < 8; ++i) {
        v16bf bfr = load_bt_frag(Kp, D_, c * 32 + tt * 16, i * 32);
        s = wmma_bf16(qf[i], bfr, s);
      }
      sc[tt] = s;
    }
    // scale + causal mask
    #pragma unroll
    for (int tt = 0; tt < 2; ++tt)
      #pragma unroll
      for (int r = 0; r < 8; ++r) {
        float v = sc[tt][r] * 0.0625f;     // 1/sqrt(256)
        int key = c * 32 + tt * 16 + n;
        int rowg = q0 + r + 8 * half;
        sc[tt][r] = (key > rowg) ? -3.0e38f : v;
      }
    // online softmax per row
    #pragma unroll
    for (int r = 0; r < 8; ++r) {
      float rm = rowmax16(fmaxf(sc[0][r], sc[1][r]));
      float nm = fmaxf(M[r], rm);
      float alpha = __expf(M[r] - nm);
      float p0 = __expf(sc[0][r] - nm);
      float p1 = __expf(sc[1][r] - nm);
      L[r] = L[r] * alpha + rowsum16(p0 + p1);
      M[r] = nm;
      #pragma unroll
      for (int t = 0; t < 16; ++t) acc[t][r] *= alpha;
      int m = r + 8 * half;
      plds[m * 32 + n]      = (__bf16)p0;
      plds[m * 32 + 16 + n] = (__bf16)p1;
    }
    asm volatile("s_wait_dscnt 0" ::: "memory");   // wave-local LDS RAW
    v16bf pf = load_a_frag(plds, 32, 0, 0);
    #pragma unroll
    for (int t = 0; t < 16; ++t) {
      v16bf vb = load_bt_frag(Vt, S_, t * 16, c * 32);
      acc[t] = wmma_bf16(pf, vb, acc[t]);
    }
  }

  int bb = bh / H_, hh = bh % H_;
  #pragma unroll
  for (int r = 0; r < 8; ++r) {
    float invL = 1.0f / L[r];
    size_t rowoff = ((size_t)bb * S_ + q0 + r + 8 * half) * INNER_ + hh * D_;
    #pragma unroll
    for (int t = 0; t < 16; ++t)
      attn[rowoff + t * 16 + n] = (__bf16)(acc[t][r] * invL);
  }
}

// ---------------------------------------------------------------------------
// 4) Wo projection + residual: x1 = attn @ Wo + x.  64 rows x 256 cols/block
// ---------------------------------------------------------------------------
__global__ __launch_bounds__(256) void k_wo(
    const __bf16* __restrict__ attn, const __bf16* __restrict__ Wot,
    const float* __restrict__ x, float* __restrict__ x1) {
  int wave = threadIdx.x >> 5;
  int lane = threadIdx.x & 31;
  int n = lane & 15, half = lane >> 4;
  int row0 = blockIdx.x * 64 + (wave >> 2) * 32;
  int col0 = (wave & 3) * 64;

  v8f acc[2][4] = {};
  gemm32x64<INNER_>(attn, INNER_, row0, Wot, col0, acc);

  #pragma unroll
  for (int g = 0; g < 2; ++g)
    #pragma unroll
    for (int t = 0; t < 4; ++t)
      #pragma unroll
      for (int r = 0; r < 8; ++r) {
        size_t idx = (size_t)(row0 + g * 16 + r + 8 * half) * D_ + col0 + 16 * t + n;
        x1[idx] = acc[g][t][r] + x[idx];
      }
}

// ---------------------------------------------------------------------------
// 5) FFN-1: mid = gelu(h2 @ fc1 + b1) -> bf16.  32 rows x 512 cols per block
// ---------------------------------------------------------------------------
__global__ __launch_bounds__(256) void k_ffn1(
    const __bf16* __restrict__ h2, const __bf16* __restrict__ f1t,
    const float* __restrict__ b1, __bf16* __restrict__ mid) {
  int wave = threadIdx.x >> 5;
  int lane = threadIdx.x & 31;
  int n = lane & 15, half = lane >> 4;
  int row0 = blockIdx.x * 32;
  int col0 = (blockIdx.y * 8 + wave) * 64;

  v8f acc[2][4] = {};
  gemm32x64<D_>(h2, D_, row0, f1t, col0, acc);

  #pragma unroll
  for (int t = 0; t < 4; ++t) {
    int col = col0 + 16 * t + n;
    float bias = b1[col];
    #pragma unroll
    for (int g = 0; g < 2; ++g)
      #pragma unroll
      for (int r = 0; r < 8; ++r) {
        float v = acc[g][t][r] + bias;
        float gl = 0.5f * v * (1.0f + erff(v * 0.70710678118f));  // exact gelu
        mid[(size_t)(row0 + g * 16 + r + 8 * half) * (4 * D_) + col] = (__bf16)gl;
      }
  }
}

// ---------------------------------------------------------------------------
// 6) FFN-2 + residual: out = mid @ fc2 + b2 + x1   (f32 output)
// ---------------------------------------------------------------------------
__global__ __launch_bounds__(256) void k_ffn2(
    const __bf16* __restrict__ mid, const __bf16* __restrict__ f2t,
    const float* __restrict__ b2, const float* __restrict__ x1,
    float* __restrict__ out) {
  int wave = threadIdx.x >> 5;
  int lane = threadIdx.x & 31;
  int n = lane & 15, half = lane >> 4;
  int row0 = blockIdx.x * 64 + (wave >> 2) * 32;
  int col0 = (wave & 3) * 64;

  v8f acc[2][4] = {};
  gemm32x64<4 * D_>(mid, 4 * D_, row0, f2t, col0, acc);

  #pragma unroll
  for (int t = 0; t < 4; ++t) {
    int col = col0 + 16 * t + n;
    float bias = b2[col];
    #pragma unroll
    for (int g = 0; g < 2; ++g)
      #pragma unroll
      for (int r = 0; r < 8; ++r) {
        size_t idx = (size_t)(row0 + g * 16 + r + 8 * half) * D_ + col;
        out[idx] = acc[g][t][r] + bias + x1[idx];
      }
  }
}

// ---------------------------------------------------------------------------
extern "C" void kernel_launch(void* const* d_in, const int* in_sizes, int n_in,
                              void* d_out, int out_size, void* d_ws, size_t ws_size,
                              hipStream_t stream) {
  const float* x    = (const float*)d_in[0];
  const float* Wq   = (const float*)d_in[1];
  const float* Wk   = (const float*)d_in[2];
  const float* Wv   = (const float*)d_in[3];
  const float* Wo   = (const float*)d_in[4];
  const float* f1w  = (const float*)d_in[5];
  const float* f1b  = (const float*)d_in[6];
  const float* f2w  = (const float*)d_in[7];
  const float* f2b  = (const float*)d_in[8];
  const float* lag  = (const float*)d_in[9];
  const float* lab  = (const float*)d_in[10];
  const float* lfg  = (const float*)d_in[11];
  const float* lfb  = (const float*)d_in[12];
  float* out = (float*)d_out;

  char* ws = (char*)d_ws;
  size_t off = 0;
  auto alloc = [&](size_t bytes) -> char* {
    char* p = ws + off;
    off += (bytes + 255) & ~(size_t)255;
    return p;
  };
  __bf16* wqt  = (__bf16*)alloc((size_t)D_ * INNER_ * 2);
  __bf16* wkt  = (__bf16*)alloc((size_t)D_ * INNER_ * 2);
  __bf16* wvt  = (__bf16*)alloc((size_t)D_ * INNER_ * 2);
  __bf16* wot  = (__bf16*)alloc((size_t)INNER_ * D_ * 2);
  __bf16* f1t  = (__bf16*)alloc((size_t)D_ * 4 * D_ * 2);
  __bf16* f2t  = (__bf16*)alloc((size_t)4 * D_ * D_ * 2);
  __bf16* hbf  = (__bf16*)alloc((size_t)BS_ * D_ * 2);
  __bf16* qb   = (__bf16*)alloc((size_t)B_ * H_ * S_ * D_ * 2);
  __bf16* kbv  = (__bf16*)alloc((size_t)B_ * H_ * S_ * D_ * 2);
  __bf16* vtb  = (__bf16*)alloc((size_t)B_ * H_ * D_ * S_ * 2);
  __bf16* attn = (__bf16*)alloc((size_t)BS_ * INNER_ * 2);
  float*  x1   = (float*) alloc((size_t)BS_ * D_ * 4);
  __bf16* h2   = (__bf16*)alloc((size_t)BS_ * D_ * 2);
  __bf16* mid  = (__bf16*)alloc((size_t)BS_ * 4 * D_ * 2);
  (void)ws_size; (void)in_sizes; (void)n_in; (void)out_size;

  // weight conversion (transposed bf16)
  auto cvt = [&](const float* s, __bf16* d, int K, int N) {
    int total = K * N;
    k_convert_transpose<<<(total + 255) / 256, 256, 0, stream>>>(s, d, K, N);
  };
  cvt(Wq,  wqt, D_, INNER_);
  cvt(Wk,  wkt, D_, INNER_);
  cvt(Wv,  wvt, D_, INNER_);
  cvt(Wo,  wot, INNER_, D_);
  cvt(f1w, f1t, D_, 4 * D_);
  cvt(f2w, f2t, 4 * D_, D_);

  // attention sublayer
  k_layernorm<<<BS_, D_, 0, stream>>>(x, lag, lab, hbf);
  k_qkv<<<dim3(BS_ / 32, INNER_ / 512, 3), 256, 0, stream>>>(
      hbf, wqt, wkt, wvt, qb, kbv, vtb);
  k_attn<<<dim3(B_ * H_, S_ / 16 / 4), 128, 0, stream>>>(qb, kbv, vtb, attn);
  k_wo<<<dim3(BS_ / 64), 256, 0, stream>>>(attn, wot, x, x1);

  // feed-forward sublayer
  k_layernorm<<<BS_, D_, 0, stream>>>(x1, lfg, lfb, h2);
  k_ffn1<<<dim3(BS_ / 32, 2), 256, 0, stream>>>(h2, f1t, f1b, mid);
  k_ffn2<<<dim3(BS_ / 64), 256, 0, stream>>>(mid, f2t, f2b, x1, out);
}
